// ContinuousAttention_20315195310356
// MI455X (gfx1250) — matrix operand order repaired and checked by reading, not compiled, not run on hardware
//
#include <hip/hip_runtime.h>
#include <hip/hip_bf16.h>
#include <math.h>

typedef __bf16 bf16;
typedef __attribute__((ext_vector_type(16))) __bf16 bf16x16;
typedef __attribute__((ext_vector_type(8)))  __bf16 bf16x8;
typedef __attribute__((ext_vector_type(8)))  float  f32x8;
typedef __attribute__((ext_vector_type(4))) unsigned int uint32x4;
typedef __attribute__((ext_vector_type(8))) int int32x8;
typedef __attribute__((ext_vector_type(4))) int int32x4;

#define WMMA_BF16(a, b, c) \
  __builtin_amdgcn_wmma_f32_16x16x32_bf16(false, (a), false, (b), (short)0, (c), false, false)

// ---- geometry ----
#define BATCH 16
#define SEQ   1024
#define CH    768
#define NHEAD 12
#define HDIM  64
#define MROWS (BATCH * SEQ)     // 16384
#define QKVN  (3 * CH)          // 2304
#define BH    (BATCH * NHEAD)   // 192

// Build 16-bit A-fragment (16x32) for one wave from a row-major row pointer.
// lanes 0-15: elems 0..15 = K kk+0..15
// lanes 16-31: elems 0..15 = K kk+16..31
__device__ __forceinline__ bf16x16 load_a_frag(const bf16* row, int kk, int base0) {
  bf16x8 lo = *(const bf16x8*)(row + kk + base0);
  bf16x8 hi = *(const bf16x8*)(row + kk + base0 + 8);
  return __builtin_shufflevector(lo, hi, 0,1,2,3,4,5,6,7,8,9,10,11,12,13,14,15);
}

// ---- Tensor Data Mover: async 2D bf16 tile (height rows x width elems) -> LDS ----
// D# packing per CDNA5 ISA 08_async_tensor §8.3/§8.4. data_size=1 (2 bytes/elem),
// tensor dims == tile dims (no OOB), groups 2/3 zero (2-D tensor).
__device__ __forceinline__ void tdm_load_2d_bf16(const void* gptr, unsigned lds_off,
                                                 unsigned width, unsigned height,
                                                 unsigned stride_elems) {
  unsigned long long ga = (unsigned long long)gptr;
  uint32x4 g0;
  g0[0] = 1u;                                               // count=1 (valid), user mode
  g0[1] = lds_off;                                          // lds_addr (bytes)
  g0[2] = (unsigned)(ga & 0xffffffffu);                     // global_addr[31:0]
  g0[3] = (unsigned)((ga >> 32) & 0x1ffffffu) | (2u << 30); // global_addr[56:32] | type=2
  int32x8 g1;
  g1[0] = (int)(1u << 16);                                  // data_size=1 (2B), wg_mask=0
  g1[1] = (int)((width & 0xffffu) << 16);                   // tensor_dim0[15:0]
  g1[2] = (int)((width >> 16) & 0xffffu)                    // tensor_dim0[31:16]
        | (int)((height & 0xffffu) << 16);                  // tensor_dim1[15:0]
  g1[3] = (int)((height >> 16) & 0xffffu)                   // tensor_dim1[31:16]
        | (int)((width & 0xffffu) << 16);                   // tile_dim0 = width
  g1[4] = (int)(height & 0xffffu);                          // tile_dim1 = height, tile_dim2=0
  g1[5] = (int)stride_elems;                                // tensor_dim0_stride[31:0]
  g1[6] = 0;                                                // stride hi / dim1_stride lo
  g1[7] = 0;
  int32x4 z4 = {0, 0, 0, 0};
  int32x8 z8 = {0, 0, 0, 0, 0, 0, 0, 0};
  __builtin_amdgcn_tensor_load_to_lds(g0, g1, z4, z4, z8, 0);
}

__device__ __forceinline__ unsigned lds_offset(const void* p) {
  // LDS aperture: flat addr low 32 bits are the LDS byte address (ISA 10.2 aperture map)
  return (unsigned)(unsigned long long)p;
}

// ---------------- stage 0: precision conversion / transposes ----------------
__global__ __launch_bounds__(256) void cvt_f32_bf16(const float* __restrict__ in,
                                                    bf16* __restrict__ out, int n) {
  int i = blockIdx.x * 256 + threadIdx.x;
  if (i < n) out[i] = (bf16)in[i];
}

// Wt[n*K + k] = W[k*N + n]   (W is KxN row-major; Wt is NxK row-major)
__global__ __launch_bounds__(256) void transpose_f32_bf16(const float* __restrict__ W,
                                                          bf16* __restrict__ Wt,
                                                          int K, int N) {
  int i = blockIdx.x * 256 + threadIdx.x;
  if (i < K * N) {
    int n = i / K, k = i % K;
    Wt[i] = (bf16)W[(size_t)k * N + n];
  }
}

// ---------------- stage 1: QKV projection ----------------
// X [16384 x 768] bf16 x WqkvT [2304 x 768] bf16 (N-major) + bqkv
// B tile (64 N-rows x 32 K) is block-shared: TDM double-buffered through LDS.
__global__ __launch_bounds__(256) void qkv_gemm(const bf16* __restrict__ X,
                                                const bf16* __restrict__ Wt,
                                                const float* __restrict__ bqkv,
                                                bf16* __restrict__ Qb,
                                                bf16* __restrict__ Kb,
                                                bf16* __restrict__ Vt) {
  __shared__ __align__(16) bf16 Bt[2][64 * 32];
  const int lane = threadIdx.x & 31;
  const int wave = threadIdx.x >> 5;
  const int m_base = blockIdx.x * 128 + wave * 16;
  const int n_base = blockIdx.y * 64;
  const int half = lane >> 4, l16 = lane & 15;
  const int base0  = half ? 16 : 0;
  const int kbase2 = half ? 16 : 0;

  const bf16* Arow = X + (size_t)(m_base + l16) * CH;
  const bf16* Bsrc = Wt + (size_t)n_base * CH;   // 64 rows, stride CH

  if (wave == 0)
    tdm_load_2d_bf16(Bsrc, lds_offset(&Bt[0][0]), 32, 64, CH);

  f32x8 acc[4] = {};
  for (int kk = 0, it = 0; kk < CH; kk += 32, ++it) {
    const int cur = it & 1;
    const bool has_next = (kk + 32) < CH;
    if (wave == 0) {
      if (has_next)
        tdm_load_2d_bf16(Bsrc + kk + 32, lds_offset(&Bt[cur ^ 1][0]), 32, 64, CH);
      if (has_next) __builtin_amdgcn_s_wait_tensorcnt(1);
      else          __builtin_amdgcn_s_wait_tensorcnt(0);
    }
    __syncthreads();
    bf16x16 a = load_a_frag(Arow, kk, base0);
#pragma unroll
    for (int t = 0; t < 4; ++t) {
      bf16x16 b = *(const bf16x16*)(&Bt[cur][(16 * t + l16) * 32 + kbase2]);
      acc[t] = WMMA_BF16(a, b, acc[t]);
    }
    __syncthreads();
  }

  const int which = n_base / CH;            // 0=Q 1=K 2=V (64 | n_base, uniform)
  const int h = (n_base % CH) / HDIM;
#pragma unroll
  for (int t = 0; t < 4; ++t) {
    const int n = n_base + 16 * t + l16;
    const float bias = bqkv[n];
    const int hd = n % HDIM;
#pragma unroll
    for (int r = 0; r < 8; ++r) {
      float v = acc[t][r] + bias;
      int m = m_base + r + 8 * half;
      int b_ = m >> 10, npos = m & 1023;
      int bh = b_ * NHEAD + h;
      bf16 bv = (bf16)v;
      if (which == 0)      Qb[((size_t)bh * SEQ + npos) * HDIM + hd] = bv;
      else if (which == 1) Kb[((size_t)bh * SEQ + npos) * HDIM + hd] = bv;
      else                 Vt[((size_t)bh * HDIM + hd) * SEQ + npos] = bv;
    }
  }
}

// ---------------- stage 2: flash attention ----------------
// 8 waves/block = 8 consecutive 16-query tiles of the SAME (b,h): K/V tiles are
// block-shared -> TDM double-buffered LDS staging. P restaged via LDS per wave.
__global__ __launch_bounds__(256) void flash_attn(const bf16* __restrict__ Qb,
                                                  const bf16* __restrict__ Kb,
                                                  const bf16* __restrict__ Vt,
                                                  bf16* __restrict__ AO) {
  __shared__ __align__(16) bf16 Psh[8][16 * 32];  // per-wave P staging
  __shared__ __align__(16) bf16 Kt[2][32 * 64];   // 32 keys x 64 hd
  __shared__ __align__(16) bf16 Vs[2][64 * 32];   // 64 hd x 32 keys
  const int lane = threadIdx.x & 31;
  const int wave = threadIdx.x >> 5;
  const int tid  = blockIdx.x * 8 + wave;
  const int bh = tid >> 6;                        // 8 | 64 -> uniform per block
  const int qt = tid & 63;
  const int half = lane >> 4, l16 = lane & 15;
  const int base0  = half ? 16 : 0;
  const int kbase2 = half ? 16 : 0;
  const int b_ = bh / NHEAD, h = bh % NHEAD;

  const bf16* Qrow = Qb + ((size_t)bh * SEQ + qt * 16 + l16) * HDIM;
  const bf16x16 aQ0 = load_a_frag(Qrow, 0, base0);
  const bf16x16 aQ1 = load_a_frag(Qrow, 32, base0);

  const bf16* Kbase = Kb + (size_t)bh * SEQ * HDIM;
  const bf16* Vbase = Vt + (size_t)bh * HDIM * SEQ;
  bf16* P = Psh[wave];

  if (wave == 0) {
    tdm_load_2d_bf16(Kbase, lds_offset(&Kt[0][0]), 64, 32, 64);      // keys 0..31
    tdm_load_2d_bf16(Vbase, lds_offset(&Vs[0][0]), 32, 64, SEQ);     // hd 0..63
  }

  float mrow[8], lrow[8];
  f32x8 accO[4] = {};
#pragma unroll
  for (int r = 0; r < 8; ++r) { mrow[r] = -3.0e38f; lrow[r] = 0.f; }

  for (int kb = 0, it = 0; kb < SEQ; kb += 32, ++it) {
    const int cur = it & 1;
    const bool has_next = (kb + 32) < SEQ;
    if (wave == 0) {
      if (has_next) {
        tdm_load_2d_bf16(Kbase + (size_t)(kb + 32) * HDIM,
                         lds_offset(&Kt[cur ^ 1][0]), 64, 32, 64);
        tdm_load_2d_bf16(Vbase + kb + 32,
                         lds_offset(&Vs[cur ^ 1][0]), 32, 64, SEQ);
      }
      if (has_next) __builtin_amdgcn_s_wait_tensorcnt(2);
      else          __builtin_amdgcn_s_wait_tensorcnt(0);
    }
    __syncthreads();

    // scores: 16 queries x 32 keys, K-dim = 64
    f32x8 sc[2];
#pragma unroll
    for (int j = 0; j < 2; ++j) {
      const bf16* Krow = &Kt[cur][(16 * j + l16) * 64 + kbase2];
      bf16x16 b0 = *(const bf16x16*)(Krow);
      bf16x16 b1 = *(const bf16x16*)(Krow + 32);
      f32x8 s = {};
      s = WMMA_BF16(aQ0, b0, s);
      s = WMMA_BF16(aQ1, b1, s);
      sc[j] = s;
    }
    // online softmax: lane-half holds rows r+8*half; reduce across 16 lanes
#pragma unroll
    for (int r = 0; r < 8; ++r) {
      float s0 = sc[0][r] * 0.125f;   // 1/sqrt(64)
      float s1 = sc[1][r] * 0.125f;
      float mx = fmaxf(s0, s1);
#pragma unroll
      for (int off = 1; off < 16; off <<= 1) mx = fmaxf(mx, __shfl_xor(mx, off, 16));
      float mnew  = fmaxf(mrow[r], mx);
      float alpha = __expf(mrow[r] - mnew);
      float p0 = __expf(s0 - mnew);
      float p1 = __expf(s1 - mnew);
      float sum = p0 + p1;
#pragma unroll
      for (int off = 1; off < 16; off <<= 1) sum += __shfl_xor(sum, off, 16);
      lrow[r] = lrow[r] * alpha + sum;
      mrow[r] = mnew;
#pragma unroll
      for (int t = 0; t < 4; ++t) accO[t][r] *= alpha;
      int row = r + 8 * half;
      P[row * 32 + l16]      = (bf16)p0;
      P[row * 32 + 16 + l16] = (bf16)p1;
    }
    asm volatile("s_wait_dscnt 0" ::: "memory");   // wave-internal LDS RAW
    bf16x16 aP = load_a_frag(P + l16 * 32, 0, base0);
    asm volatile("" ::: "memory");
#pragma unroll
    for (int t = 0; t < 4; ++t) {
      bf16x16 bV = *(const bf16x16*)(&Vs[cur][(16 * t + l16) * 32 + kbase2]);
      accO[t] = WMMA_BF16(aP, bV, accO[t]);
    }
    __syncthreads();
  }

  // epilogue: divide by l, write bf16 into [MROWS, CH] at head columns
  const size_t mbase = (size_t)b_ * SEQ + qt * 16;
#pragma unroll
  for (int t = 0; t < 4; ++t) {
    int col = h * HDIM + 16 * t + l16;
#pragma unroll
    for (int r = 0; r < 8; ++r) {
      size_t mm = mbase + r + 8 * half;
      AO[mm * CH + col] = (bf16)(accO[t][r] / lrow[r]);
    }
  }
}

// ---------------- stage 3: output projection ----------------
__global__ __launch_bounds__(256) void out_proj(const bf16* __restrict__ A,
                                                const bf16* __restrict__ Wt,
                                                const float* __restrict__ bias,
                                                float* __restrict__ Out) {
  __shared__ __align__(16) bf16 Bt[2][64 * 32];
  const int lane = threadIdx.x & 31;
  const int wave = threadIdx.x >> 5;
  const int m_base = blockIdx.x * 128 + wave * 16;
  const int n_base = blockIdx.y * 64;
  const int half = lane >> 4, l16 = lane & 15;
  const int base0  = half ? 16 : 0;
  const int kbase2 = half ? 16 : 0;

  const bf16* Arow = A + (size_t)(m_base + l16) * CH;
  const bf16* Bsrc = Wt + (size_t)n_base * CH;

  if (wave == 0)
    tdm_load_2d_bf16(Bsrc, lds_offset(&Bt[0][0]), 32, 64, CH);

  f32x8 acc[4] = {};
  for (int kk = 0, it = 0; kk < CH; kk += 32, ++it) {
    const int cur = it & 1;
    const bool has_next = (kk + 32) < CH;
    if (wave == 0) {
      if (has_next)
        tdm_load_2d_bf16(Bsrc + kk + 32, lds_offset(&Bt[cur ^ 1][0]), 32, 64, CH);
      if (has_next) __builtin_amdgcn_s_wait_tensorcnt(1);
      else          __builtin_amdgcn_s_wait_tensorcnt(0);
    }
    __syncthreads();
    bf16x16 a = load_a_frag(Arow, kk, base0);
#pragma unroll
    for (int t = 0; t < 4; ++t) {
      bf16x16 b = *(const bf16x16*)(&Bt[cur][(16 * t + l16) * 32 + kbase2]);
      acc[t] = WMMA_BF16(a, b, acc[t]);
    }
    __syncthreads();
  }
#pragma unroll
  for (int t = 0; t < 4; ++t) {
    int n = n_base + 16 * t + l16;
    float bb = bias[n];
#pragma unroll
    for (int r = 0; r < 8; ++r) {
      size_t m = (size_t)m_base + r + 8 * half;
      Out[m * CH + n] = acc[t][r] + bb;
    }
  }
}

// ---------------- host launcher ----------------
extern "C" void kernel_launch(void* const* d_in, const int* in_sizes, int n_in,
                              void* d_out, int out_size, void* d_ws, size_t ws_size,
                              hipStream_t stream) {
  const float* x    = (const float*)d_in[0];
  const float* Wqkv = (const float*)d_in[1];
  const float* bqkv = (const float*)d_in[2];
  const float* Wout = (const float*)d_in[3];
  const float* bout = (const float*)d_in[4];
  float* out = (float*)d_out;

  char* ws = (char*)d_ws;
  size_t o = 0;
  bf16* Xb  = (bf16*)(ws + o); o += (size_t)MROWS * CH * 2;
  bf16* WqT = (bf16*)(ws + o); o += (size_t)QKVN * CH * 2;
  bf16* WoT = (bf16*)(ws + o); o += (size_t)CH * CH * 2;
  bf16* Qb  = (bf16*)(ws + o); o += (size_t)BH * SEQ * HDIM * 2;
  bf16* Kb  = (bf16*)(ws + o); o += (size_t)BH * SEQ * HDIM * 2;
  bf16* Vt  = (bf16*)(ws + o); o += (size_t)BH * HDIM * SEQ * 2;
  bf16* AO  = (bf16*)(ws + o); o += (size_t)MROWS * CH * 2;

  {
    int n = MROWS * CH;
    cvt_f32_bf16<<<(n + 255) / 256, 256, 0, stream>>>(x, Xb, n);
  }
  {
    int n = CH * QKVN;
    transpose_f32_bf16<<<(n + 255) / 256, 256, 0, stream>>>(Wqkv, WqT, CH, QKVN);
  }
  {
    int n = CH * CH;
    transpose_f32_bf16<<<(n + 255) / 256, 256, 0, stream>>>(Wout, WoT, CH, CH);
  }
  qkv_gemm<<<dim3(MROWS / 128, QKVN / 64), 256, 0, stream>>>(Xb, WqT, bqkv, Qb, Kb, Vt);
  flash_attn<<<(BH * (SEQ / 16)) / 8, 256, 0, stream>>>(Qb, Kb, Vt, AO);
  out_proj<<<dim3(MROWS / 128, CH / 64), 256, 0, stream>>>(AO, WoT, bout, out);
}
